// MultiHeadAttention_55611236548678
// MI455X (gfx1250) — compile-verified, matrix operand order checked by reading, not run on hardware
//
#include <hip/hip_runtime.h>
#include <hip/hip_bf16.h>
#include <cstdint>

// MHA: B=2, S=2048, D_MODEL=1024, H=16, d=64
#define Bn 2
#define Sn 2048
#define DM 1024
#define Hn 16
#define Dh 64

typedef __attribute__((ext_vector_type(16))) _Float16 v16h;
typedef __attribute__((ext_vector_type(8)))  _Float16 v8h;
typedef __attribute__((ext_vector_type(4)))  _Float16 v4h;
typedef __attribute__((ext_vector_type(8)))  float    v8f;
typedef __attribute__((ext_vector_type(4)))  float    v4f;

// ISA 7.12.2 16-bit fragment K-map: elements 0..7 -> K = hi*8 + (0..7),
// elements 8..15 -> K = hi*8 + 16 + (0..7). So a fragment whose elements are
// contiguous along the source K-axis is exactly two 16-byte loads per lane.
__device__ __forceinline__ v16h frag2(v8h a, v8h b) {
    return __builtin_shufflevector(a, b, 0, 1, 2, 3, 4, 5, 6, 7,
                                   8, 9, 10, 11, 12, 13, 14, 15);
}

// base must already include the lane's row/col offset AND + hi*8 halves.
__device__ __forceinline__ v16h load_frag16(const _Float16* base) {
    const v8h* p = (const v8h*)base;
    return frag2(p[0], p[2]);  // +0 and +16 halves
}

__device__ __forceinline__ v8f wmma_f16(v16h a, v16h b, v8f c) {
    return __builtin_amdgcn_wmma_f32_16x16x32_f16(
        /*neg_a=*/false, a, /*neg_b=*/false, b,
        /*c_mod=*/(short)0, c, /*reuse_a=*/false, /*reuse_b=*/false);
}

// ---------------------------------------------------------------------------
// Kernel A: per-head projection  X[4096,1024] x W[h][1024,64] + bias -> f16
//   TR=0: out[B,H,S,64] (Q,K);  TR=1: out[B,H,64,S] (V, transposed for P@V)
// grid (32, 16), block 256 (8 waves); tile 128(M) x 64(N), K slab = 32
// ---------------------------------------------------------------------------
template <bool TR>
__global__ void __launch_bounds__(256)
qkv_kernel(const float* __restrict__ X, const float* __restrict__ W,
           const float* __restrict__ bias, _Float16* __restrict__ out) {
    __shared__ _Float16 Xs[128][40];  // X slab, row-major (pitch 80B)
    __shared__ _Float16 Wt[64][40];   // W slab TRANSPOSED: Wt[n][k]
    const int tid = threadIdx.x, wave = tid >> 5, lane = tid & 31;
    const int rowTile = blockIdx.x, h = blockIdx.y;
    const int m0 = wave * 16;
    const int hi = lane >> 4, nidx = lane & 15;
    v8f c[4] = {};

    for (int k0 = 0; k0 < DM; k0 += 32) {
        __syncthreads();
        // stage X: 128x32 f32 -> f16, vectorized float4 -> v4h
        for (int i = tid; i < 1024; i += 256) {
            int r = i >> 3, cc = (i & 7) * 4;
            v4f x = *(const v4f*)&X[(size_t)(rowTile * 128 + r) * DM + k0 + cc];
            v4h xh = {(_Float16)x.x, (_Float16)x.y, (_Float16)x.z, (_Float16)x.w};
            *(v4h*)&Xs[r][cc] = xh;
        }
        // stage W transposed: Wt[n][k]
        for (int i = tid; i < 512; i += 256) {
            int r = i >> 4, cc = (i & 15) * 4;  // r = k row, cc = n col
            v4f w = *(const v4f*)&W[((size_t)h * DM + k0 + r) * Dh + cc];
            Wt[cc + 0][r] = (_Float16)w.x;
            Wt[cc + 1][r] = (_Float16)w.y;
            Wt[cc + 2][r] = (_Float16)w.z;
            Wt[cc + 3][r] = (_Float16)w.w;
        }
        if (k0 + 32 < DM)
            __builtin_prefetch(&X[(size_t)(rowTile * 128 + (tid >> 1)) * DM + k0 + 32], 0, 1);
        __syncthreads();

        v16h a = load_frag16(&Xs[m0 + nidx][hi * 8]);
#pragma unroll
        for (int nt = 0; nt < 4; ++nt) {
            v16h bf = load_frag16(&Wt[nt * 16 + nidx][hi * 8]);
            c[nt] = wmma_f16(a, bf, c[nt]);
        }
    }

#pragma unroll
    for (int nt = 0; nt < 4; ++nt) {
        int d = nt * 16 + nidx;
        float bv = bias[h * Dh + d];
#pragma unroll
        for (int r = 0; r < 8; ++r) {
            int rowg = rowTile * 128 + m0 + r + 8 * hi;
            int bb = rowg >> 11, s = rowg & (Sn - 1);
            _Float16 val = (_Float16)(c[nt][r] + bv);
            if (TR)
                out[(((size_t)bb * Hn + h) * Dh + d) * Sn + s] = val;
            else
                out[(((size_t)bb * Hn + h) * Sn + s) * Dh + d] = val;
        }
    }
}

// ---------------------------------------------------------------------------
// Kernel B: attention for one (b, h, 16-row q tile). grid = 4096, block 256.
//  Phase1: scores = Q K^T / 8 -> f16 LDS [16][2048]
//  Phase2: mask (vectorized byte stream) + softmax over k;
//          write probs to attn_out [H,Sk,B,Sq] (q contiguous)
//  Phase3: O = P V (V transposed [B,H,d,S]), K split over 8 waves
// ---------------------------------------------------------------------------
__global__ void __launch_bounds__(256)
attn_kernel(const _Float16* __restrict__ Qh, const _Float16* __restrict__ Kh,
            const _Float16* __restrict__ Vt, const unsigned char* __restrict__ mask,
            float* __restrict__ attn_out, _Float16* __restrict__ Oh) {
    __shared__ _Float16 sc[16][Sn];   // 64 KB score/prob tile
    __shared__ _Float16 Qs[16][Dh];   // 2 KB
    __shared__ float red[16][16];
    __shared__ float rowmax[16], rowsum[16];
    __shared__ float part[8][16][Dh]; // 32 KB wave partials for P@V

    const int tid = threadIdx.x, wave = tid >> 5, lane = tid & 31;
    const int qt = blockIdx.x & 127;
    const int h  = (blockIdx.x >> 7) & 15;
    const int b  = blockIdx.x >> 11;
    const size_t headBase = ((size_t)b * Hn + h) * Sn * Dh;  // same for K and Vt
    const int q0 = qt * 16;
    const int hi = lane >> 4, nidx = lane & 15;

    for (int i = tid; i < 16 * Dh / 8; i += 256) {  // 128 v8h chunks
        int q = i >> 3, d = (i & 7) * 8;
        *(v8h*)&Qs[q][d] = *(const v8h*)&Qh[headBase + (size_t)(q0 + q) * Dh + d];
    }
    __syncthreads();

    // ---- Phase 1: raw scores (mask applied in phase 2 on contiguous chunks) ----
    v16h a0 = load_frag16(&Qs[nidx][hi * 8]);
    v16h a1 = load_frag16(&Qs[nidx][32 + hi * 8]);
    for (int t = 0; t < 16; ++t) {
        int kb = wave * 256 + t * 16;
        // B[d][k] = K[kb+n][d]; K row contiguous in d -> 4 x b128 per lane
        const v8h* kp = (const v8h*)&Kh[headBase + (size_t)(kb + nidx) * Dh + hi * 8];
        v16h b0 = frag2(kp[0], kp[2]);
        v16h b1 = frag2(kp[4], kp[6]);
        v8f c = {};
        c = wmma_f16(a0, b0, c);
        c = wmma_f16(a1, b1, c);
#pragma unroll
        for (int r = 0; r < 8; ++r)
            sc[r + 8 * hi][kb + nidx] = (_Float16)(c[r] * 0.125f);  // 1/sqrt(64)
    }
    __syncthreads();

    // ---- Phase 2: mask + softmax; thread (row, sub) owns contiguous 128 keys ----
    const int row = tid >> 4, sub = tid & 15;
    const int kbase = sub * 128;
    const unsigned char* mrow = mask + (size_t)(q0 + row) * Sn + kbase;

    // masked max pass: mask bytes are contiguous -> uint2 (8B) loads, coalesced
    float lm = -3.0e38f;
    for (int cchunk = 0; cchunk < 16; ++cchunk) {
        v8h v = *(v8h*)&sc[row][kbase + cchunk * 8];
        uint2 mb = *(const uint2*)(mrow + cchunk * 8);
        unsigned long long m64 = ((unsigned long long)mb.y << 32) | (unsigned long long)mb.x;
#pragma unroll
        for (int j = 0; j < 8; ++j) {
            if (((m64 >> (8 * j)) & 0xffu) == 0u)
                v[j] = (_Float16)(-60000.0f);  // finite in f16: matches -1e9 semantics
            lm = fmaxf(lm, (float)v[j]);
        }
        *(v8h*)&sc[row][kbase + cchunk * 8] = v;
    }
    red[row][sub] = lm;
    __syncthreads();
    if (sub == 0) {
        float m = red[row][0];
        for (int i = 1; i < 16; ++i) m = fmaxf(m, red[row][i]);
        rowmax[row] = m;
    }
    __syncthreads();
    float rm = rowmax[row], ls = 0.f;
    for (int cchunk = 0; cchunk < 16; ++cchunk) {
        v8h v = *(v8h*)&sc[row][kbase + cchunk * 8];
#pragma unroll
        for (int j = 0; j < 8; ++j) {
            float e = __expf((float)v[j] - rm);
            v[j] = (_Float16)e;  // e in [0,1]: f16-safe
            ls += e;
        }
        *(v8h*)&sc[row][kbase + cchunk * 8] = v;
    }
    red[row][sub] = ls;
    __syncthreads();
    if (sub == 0) {
        float s = 0.f;
        for (int i = 0; i < 16; ++i) s += red[row][i];
        rowsum[row] = s;
    }
    __syncthreads();
    const float inv = 1.0f / rowsum[row];
    for (int cchunk = 0; cchunk < 16; ++cchunk) {
        v8h v = *(v8h*)&sc[row][kbase + cchunk * 8];
#pragma unroll
        for (int j = 0; j < 8; ++j) v[j] = (_Float16)((float)v[j] * inv);
        *(v8h*)&sc[row][kbase + cchunk * 8] = v;
    }
    __syncthreads();

    // ---- write attn weights [H, Sk, B, Sq]; consecutive tids -> consecutive q ----
    {
        int qw = tid & 15, kw = tid >> 4;
        for (int i = 0; i < 128; ++i) {
            int k = i * 16 + kw;
            attn_out[(((size_t)h * Sn + k) * Bn + b) * Sn + q0 + qw] = (float)sc[qw][k];
        }
    }

    // ---- Phase 3: P @ V, K split across waves; Vt[b,h,d,s] contiguous in k ----
    v8f c4[4] = {};
    for (int ks = 0; ks < 8; ++ks) {
        int kb = wave * 256 + ks * 32;
        v16h a = load_frag16(&sc[nidx][kb + hi * 8]);
#pragma unroll
        for (int nt = 0; nt < 4; ++nt) {
            // B[k][d]: lane = d column, elements contiguous along k in Vt
            v16h bf = load_frag16(&Vt[headBase + (size_t)(nt * 16 + nidx) * Sn + kb + hi * 8]);
            c4[nt] = wmma_f16(a, bf, c4[nt]);
        }
    }
#pragma unroll
    for (int nt = 0; nt < 4; ++nt)
#pragma unroll
        for (int r = 0; r < 8; ++r)
            part[wave][r + 8 * hi][nt * 16 + nidx] = c4[nt][r];
    __syncthreads();
    for (int e = tid; e < 16 * Dh; e += 256) {
        int q = e >> 6, d = e & 63;
        float s = 0.f;
#pragma unroll
        for (int w = 0; w < 8; ++w) s += part[w][q][d];
        Oh[((size_t)b * Sn + q0 + q) * DM + h * Dh + d] = (_Float16)s;
    }
}

// ---------------------------------------------------------------------------
// Kernel C: out = concat[4096,1024](f16) x Wo[1024,1024] + bo -> f32
// grid (32, 16), block 256; tile 128 x 64, K slab 32
// ---------------------------------------------------------------------------
__global__ void __launch_bounds__(256)
proj_kernel(const _Float16* __restrict__ Oh, const float* __restrict__ Wo,
            const float* __restrict__ bo, float* __restrict__ out) {
    __shared__ _Float16 As[128][40];
    __shared__ _Float16 Bt[64][40];   // Wo slab TRANSPOSED: Bt[n][k]
    const int tid = threadIdx.x, wave = tid >> 5, lane = tid & 31;
    const int rowTile = blockIdx.x, n0 = blockIdx.y * 64;
    const int m0 = wave * 16;
    const int hi = lane >> 4, nidx = lane & 15;
    v8f c[4] = {};

    for (int k0 = 0; k0 < DM; k0 += 32) {
        __syncthreads();
        // stage A: f16 copy, v8h chunks
        for (int i = tid; i < 512; i += 256) {
            int r = i >> 2, cc = (i & 3) * 8;
            *(v8h*)&As[r][cc] = *(const v8h*)&Oh[(size_t)(rowTile * 128 + r) * DM + k0 + cc];
        }
        // stage Wo transposed
        for (int i = tid; i < 512; i += 256) {
            int r = i >> 4, cc = (i & 15) * 4;
            v4f w = *(const v4f*)&Wo[(size_t)(k0 + r) * DM + n0 + cc];
            Bt[cc + 0][r] = (_Float16)w.x;
            Bt[cc + 1][r] = (_Float16)w.y;
            Bt[cc + 2][r] = (_Float16)w.z;
            Bt[cc + 3][r] = (_Float16)w.w;
        }
        __syncthreads();

        v16h a = load_frag16(&As[m0 + nidx][hi * 8]);
#pragma unroll
        for (int nt = 0; nt < 4; ++nt) {
            v16h bf = load_frag16(&Bt[nt * 16 + nidx][hi * 8]);
            c[nt] = wmma_f16(a, bf, c[nt]);
        }
    }

#pragma unroll
    for (int nt = 0; nt < 4; ++nt) {
        int d = n0 + nt * 16 + nidx;
        float bb = bo[d];
#pragma unroll
        for (int r = 0; r < 8; ++r) {
            int rowg = rowTile * 128 + m0 + r + 8 * hi;
            out[(size_t)rowg * DM + d] = c[nt][r] + bb;
        }
    }
}

extern "C" void kernel_launch(void* const* d_in, const int* in_sizes, int n_in,
                              void* d_out, int out_size, void* d_ws, size_t ws_size,
                              hipStream_t stream) {
    (void)in_sizes; (void)n_in; (void)out_size; (void)ws_size;
    const float* pre_q = (const float*)d_in[0];
    const float* pre_k = (const float*)d_in[1];
    const float* pre_v = (const float*)d_in[2];
    const unsigned char* mask = (const unsigned char*)d_in[3];
    const float* Wq = (const float*)d_in[4];
    const float* bq = (const float*)d_in[5];
    const float* Wk = (const float*)d_in[6];
    const float* bk = (const float*)d_in[7];
    const float* Wv = (const float*)d_in[8];
    const float* bv = (const float*)d_in[9];
    const float* Wo = (const float*)d_in[10];
    const float* bo = (const float*)d_in[11];

    float* out  = (float*)d_out;
    float* attn = out + (size_t)Bn * Sn * DM;  // outputs concatenated flat

    const size_t headElems = (size_t)Bn * Hn * Sn * Dh;  // 4,194,304
    _Float16* Qh = (_Float16*)d_ws;
    _Float16* Kh = Qh + headElems;
    _Float16* Vt = Kh + headElems;  // V stored [B,H,d,S]
    _Float16* Oh = Vt + headElems;  // concat layout [B,S,H*d]; 32 MB ws total

    dim3 gProj(32, 16);
    qkv_kernel<false><<<gProj, 256, 0, stream>>>(pre_q, Wq, bq, Qh);
    qkv_kernel<false><<<gProj, 256, 0, stream>>>(pre_k, Wk, bk, Kh);
    qkv_kernel<true ><<<gProj, 256, 0, stream>>>(pre_v, Wv, bv, Vt);

    attn_kernel<<<Bn * Hn * (Sn / 16), 256, 0, stream>>>(Qh, Kh, Vt, mask, attn, Oh);

    proj_kernel<<<dim3(32, 16), 256, 0, stream>>>(Oh, Wo, bo, out);
}
